// TemporalMambaSSM_11012296147045
// MI455X (gfx1250) — compile-verified, matrix operand order checked by reading, not run on hardware
//
#include <hip/hip_runtime.h>
#include <hip/hip_bf16.h>
#include <cstdint>

#define D_MODEL 256
#define D_INNER 512
#define D_STATE 16
#define DT_RANK 16
#define SEQ_N   128
#define BATCH   392
#define M_TOTAL (BATCH * SEQ_N)   // 50176 rows, divisible by 16

typedef __attribute__((ext_vector_type(16))) __bf16 v16bf;
typedef __attribute__((ext_vector_type(8)))  __bf16 v8bf;
typedef __attribute__((ext_vector_type(8)))  float  v8f;

// ---------------------------------------------------------------------------
// Build a 16x32 bf16 WMMA operand fragment from a row-major bf16 source.
// CDNA5 16-bit A layout (ISA 7.12.2): lanes 0-15 hold rows r0..r0+15 with
// K-chunks {k0..k0+7} and {k0+16..k0+23}; lanes 16-31 hold the same rows
// with K-chunks {k0+8..k0+15} and {k0+24..k0+31}. B mirrors with N for M,
// and since B = W^T with W row-major over K, both operands are two
// contiguous 16-byte chunks per lane -> two global_load_b128, no VALU.
// ---------------------------------------------------------------------------
__device__ __forceinline__ v16bf load_frag_bf16(const __bf16* __restrict__ base,
                                                int ld, int r0, int k0, int lane) {
  const int r  = lane & 15;
  const int ko = (lane >> 4) << 3;
  const __bf16* p = base + (size_t)(r0 + r) * (size_t)ld + (size_t)(k0 + ko);
  const v8bf lo = *(const v8bf*)p;          // 16B aligned: ld, k0+ko multiples of 8
  const v8bf hi = *(const v8bf*)(p + 16);
  return __builtin_shufflevector(lo, hi, 0, 1, 2, 3, 4, 5, 6, 7,
                                 8, 9, 10, 11, 12, 13, 14, 15);
}

// ---------------------------------------------------------------------------
// K0: one-shot fp32 -> bf16 conversion (weights + x). Memory bound.
// ---------------------------------------------------------------------------
__global__ __launch_bounds__(256) void k_cvt_bf16(const float* __restrict__ in,
                                                  __bf16* __restrict__ out) {
  const size_t i = (size_t)blockIdx.x * 256 + threadIdx.x;  // exact grids only
  out[i] = (__bf16)in[i];
}

// ---------------------------------------------------------------------------
// K1: xz[M,1024] = x[M,256] @ in_proj_w[1024,256]^T   (bf16 in, f32 out)
// 8 waves/block; each wave owns a 16x64 strip: 1 A-frag drives 4 WMMAs.
// ---------------------------------------------------------------------------
__global__ __launch_bounds__(256) void k_in_proj(const __bf16* __restrict__ x,
                                                 const __bf16* __restrict__ w,
                                                 float* __restrict__ xz) {
  const int lane = threadIdx.x & 31;
  const int wv   = threadIdx.x >> 5;
  const int r0   = blockIdx.y * 16;
  const int c0   = blockIdx.x * 512 + wv * 64;

  v8f acc0 = {}, acc1 = {}, acc2 = {}, acc3 = {};
  for (int k = 0; k < D_MODEL; k += 32) {
    if (k + 32 < D_MODEL)
      __builtin_prefetch(x + (size_t)(r0 + (lane & 15)) * D_MODEL + k + 32, 0, 0);
    const v16bf a  = load_frag_bf16(x, D_MODEL, r0, k, lane);
    const v16bf b0 = load_frag_bf16(w, D_MODEL, c0 +  0, k, lane);
    const v16bf b1 = load_frag_bf16(w, D_MODEL, c0 + 16, k, lane);
    const v16bf b2 = load_frag_bf16(w, D_MODEL, c0 + 32, k, lane);
    const v16bf b3 = load_frag_bf16(w, D_MODEL, c0 + 48, k, lane);
    acc0 = __builtin_amdgcn_wmma_f32_16x16x32_bf16(false, a, false, b0, (short)0, acc0, false, false);
    acc1 = __builtin_amdgcn_wmma_f32_16x16x32_bf16(false, a, false, b1, (short)0, acc1, false, false);
    acc2 = __builtin_amdgcn_wmma_f32_16x16x32_bf16(false, a, false, b2, (short)0, acc2, false, false);
    acc3 = __builtin_amdgcn_wmma_f32_16x16x32_bf16(false, a, false, b3, (short)0, acc3, false, false);
  }
  // C/D layout: VGPR i -> row i (lanes 0-15) / i+8 (lanes 16-31), col = lane&15
  const int n  = lane & 15;
  const int mo = (lane >> 4) << 3;
#pragma unroll
  for (int i = 0; i < 8; ++i) {
    float* row = xz + (size_t)(r0 + mo + i) * (2 * D_INNER) + c0 + n;
    row[0]  = acc0[i];
    row[16] = acc1[i];
    row[32] = acc2[i];
    row[48] = acc3[i];
  }
}

// ---------------------------------------------------------------------------
// K2: depthwise causal conv(4) + bias + SiLU -> u (f32 for scan, bf16 for GEMM)
// ---------------------------------------------------------------------------
__global__ __launch_bounds__(256) void k_conv_silu(const float* __restrict__ xz,
                                                   const float* __restrict__ cw,
                                                   const float* __restrict__ cb,
                                                   float* __restrict__ u,
                                                   __bf16* __restrict__ ubf) {
  const size_t idx = (size_t)blockIdx.x * 256 + threadIdx.x;  // over M*512
  const int    d   = (int)(idx & (D_INNER - 1));
  const size_t m   = idx >> 9;
  const int    n   = (int)(m & (SEQ_N - 1));
  float acc = cb[d];
#pragma unroll
  for (int j = 0; j < 4; ++j) {
    const int nj = n - 3 + j;
    if (nj >= 0) acc += cw[d * 4 + j] * xz[(m - 3 + j) * (2 * D_INNER) + d];
  }
  const float sig = 1.f / (1.f + __expf(-acc));
  const float uv  = acc * sig;
  u[idx]   = uv;
  ubf[idx] = (__bf16)uv;
}

// ---------------------------------------------------------------------------
// K3: x_dbl[M,48] = u[M,512] @ x_proj_w[48,512]^T ; 3 waves = 3 col tiles
// ---------------------------------------------------------------------------
__global__ __launch_bounds__(96) void k_x_proj(const __bf16* __restrict__ u,
                                               const __bf16* __restrict__ w,
                                               float* __restrict__ xdbl) {
  const int lane = threadIdx.x & 31;
  const int wv   = threadIdx.x >> 5;
  const int r0   = blockIdx.x * 16;
  const int c0   = wv * 16;
  v8f acc = {};
  for (int k = 0; k < D_INNER; k += 32) {
    if (k + 32 < D_INNER)
      __builtin_prefetch(u + (size_t)(r0 + (lane & 15)) * D_INNER + k + 32, 0, 0);
    const v16bf a = load_frag_bf16(u, D_INNER, r0, k, lane);
    const v16bf b = load_frag_bf16(w, D_INNER, c0, k, lane);
    acc = __builtin_amdgcn_wmma_f32_16x16x32_bf16(false, a, false, b, (short)0, acc, false, false);
  }
  const int n  = lane & 15;
  const int mo = (lane >> 4) << 3;
#pragma unroll
  for (int i = 0; i < 8; ++i)
    xdbl[(size_t)(r0 + mo + i) * (DT_RANK + 2 * D_STATE) + c0 + n] = acc[i];
}

// ---------------------------------------------------------------------------
// K4: delta[M,512] = softplus(x_dbl[:, :16] @ dt_proj_w^T + dt_proj_b)
// K=16 -> memory bound; VALU with L2-broadcast x_dbl row reads
// ---------------------------------------------------------------------------
__global__ __launch_bounds__(256) void k_dt(const float* __restrict__ xdbl,
                                            const float* __restrict__ dtw,
                                            const float* __restrict__ dtb,
                                            float* __restrict__ delta) {
  const size_t idx = (size_t)blockIdx.x * 256 + threadIdx.x;  // over M*512
  const int    d   = (int)(idx & (D_INNER - 1));
  const size_t m   = idx >> 9;
  float acc = dtb[d];
  const float* xr = xdbl + m * 48;
#pragma unroll
  for (int r = 0; r < DT_RANK; ++r) acc += xr[r] * dtw[d * DT_RANK + r];
  delta[idx] = (acc > 20.f) ? acc : log1pf(__expf(acc));
}

// ---------------------------------------------------------------------------
// K5: selective scan. One thread per (b,d): h[16] register-resident, 128
// sequential steps. y = (scan_y + u*Dp) * silu(z), stored bf16 for out_proj.
// ---------------------------------------------------------------------------
__global__ __launch_bounds__(256) void k_scan(const float* __restrict__ xz,
                                              const float* __restrict__ u,
                                              const float* __restrict__ xdbl,
                                              const float* __restrict__ delta,
                                              const float* __restrict__ A_log,
                                              const float* __restrict__ Dp,
                                              __bf16* __restrict__ y) {
  const int b = blockIdx.x >> 1;
  const int d = ((blockIdx.x & 1) << 8) + threadIdx.x;
  float Arow[D_STATE];
#pragma unroll
  for (int s = 0; s < D_STATE; ++s) Arow[s] = -__expf(A_log[d * D_STATE + s]);
  const float dpv = Dp[d];
  float h[D_STATE];
#pragma unroll
  for (int s = 0; s < D_STATE; ++s) h[s] = 0.f;

  for (int n = 0; n < SEQ_N; ++n) {
    const size_t m   = (size_t)b * SEQ_N + n;
    const float  dlt = delta[m * D_INNER + d];
    const float  ut  = u[m * D_INNER + d];
    const float  zt  = xz[m * (2 * D_INNER) + D_INNER + d];
    const float  du  = dlt * ut;
    const float* bc  = xdbl + m * 48;
    float yt = 0.f;
#pragma unroll
    for (int s = 0; s < D_STATE; ++s) {
      const float dA = __expf(dlt * Arow[s]);
      h[s] = dA * h[s] + du * bc[DT_RANK + s];
      yt += h[s] * bc[DT_RANK + D_STATE + s];
    }
    float yv = yt + ut * dpv;
    const float sig = 1.f / (1.f + __expf(-zt));
    yv *= zt * sig;  // * silu(z)
    y[m * D_INNER + d] = (__bf16)yv;
  }
}

// ---------------------------------------------------------------------------
// K6: out = LayerNorm( y[M,512] @ out_proj_w[256,512]^T ), fully fused.
// Block owns 16 full rows (16x256): 8 waves x two 16x16 tiles, K=512, then
// LDS-staged mean/var reduction + normalize.
// ---------------------------------------------------------------------------
__global__ __launch_bounds__(256) void k_out_proj_ln(const __bf16* __restrict__ y,
                                                     const __bf16* __restrict__ w,
                                                     const float* __restrict__ lng,
                                                     const float* __restrict__ lnb,
                                                     float* __restrict__ out) {
  __shared__ float tile[16 * D_MODEL];
  __shared__ float red1[16][16];
  __shared__ float red2[16][16];

  const int lane = threadIdx.x & 31;
  const int wv   = threadIdx.x >> 5;
  const int r0   = blockIdx.x * 16;
  const int c0   = wv * 32;

  v8f c0acc = {}, c1acc = {};
  for (int k = 0; k < D_INNER; k += 32) {
    if (k + 32 < D_INNER)
      __builtin_prefetch(y + (size_t)(r0 + (lane & 15)) * D_INNER + k + 32, 0, 0);
    const v16bf a  = load_frag_bf16(y, D_INNER, r0, k, lane);
    const v16bf b0 = load_frag_bf16(w, D_INNER, c0, k, lane);
    const v16bf b1 = load_frag_bf16(w, D_INNER, c0 + 16, k, lane);
    c0acc = __builtin_amdgcn_wmma_f32_16x16x32_bf16(false, a, false, b0, (short)0, c0acc, false, false);
    c1acc = __builtin_amdgcn_wmma_f32_16x16x32_bf16(false, a, false, b1, (short)0, c1acc, false, false);
  }
  const int n  = lane & 15;
  const int mo = (lane >> 4) << 3;
#pragma unroll
  for (int i = 0; i < 8; ++i) {
    tile[(mo + i) * D_MODEL + c0 + n]      = c0acc[i];
    tile[(mo + i) * D_MODEL + c0 + 16 + n] = c1acc[i];
  }
  __syncthreads();

  // LayerNorm over 256 cols per row; 16 threads per row reduce 16 elems each.
  const int row = threadIdx.x >> 4;
  const int seg = threadIdx.x & 15;
  float s = 0.f, s2 = 0.f;
#pragma unroll
  for (int j = 0; j < 16; ++j) {
    const float v = tile[row * D_MODEL + seg * 16 + j];
    s += v; s2 += v * v;
  }
  red1[row][seg] = s;
  red2[row][seg] = s2;
  __syncthreads();

  float mean = 0.f, msq = 0.f;
#pragma unroll
  for (int j = 0; j < 16; ++j) { mean += red1[row][j]; msq += red2[row][j]; }
  mean *= (1.f / 256.f);
  msq  *= (1.f / 256.f);
  const float rstd = rsqrtf(msq - mean * mean + 1e-5f);
#pragma unroll
  for (int j = 0; j < 16; ++j) {
    const int   cc = seg * 16 + j;
    const float v  = tile[row * D_MODEL + cc];
    out[(size_t)(r0 + row) * D_MODEL + cc] = (v - mean) * rstd * lng[cc] + lnb[cc];
  }
}

// ---------------------------------------------------------------------------
extern "C" void kernel_launch(void* const* d_in, const int* in_sizes, int n_in,
                              void* d_out, int out_size, void* d_ws, size_t ws_size,
                              hipStream_t stream) {
  (void)in_sizes; (void)n_in; (void)out_size; (void)ws_size;
  const float* x    = (const float*)d_in[0];
  const float* inw  = (const float*)d_in[1];   // (1024, 256)
  const float* cw   = (const float*)d_in[2];   // (512, 1, 4)
  const float* cb   = (const float*)d_in[3];   // (512,)
  const float* xpw  = (const float*)d_in[4];   // (48, 512)
  const float* dtw  = (const float*)d_in[5];   // (512, 16)
  const float* dtb  = (const float*)d_in[6];   // (512,)
  const float* alog = (const float*)d_in[7];   // (512, 16)
  const float* dp   = (const float*)d_in[8];   // (512,)
  const float* ow   = (const float*)d_in[9];   // (256, 512)
  const float* lng  = (const float*)d_in[10];  // (256,)
  const float* lnb  = (const float*)d_in[11];  // (256,)
  float* out = (float*)d_out;

  // --- workspace partition (all chunks 16B-aligned) ---
  char* p = (char*)d_ws;
  float* xz = (float*)p;        p += (size_t)M_TOTAL * 1024 * sizeof(float);
  float* u  = (float*)p;        p += (size_t)M_TOTAL * D_INNER * sizeof(float);
  float* xd = (float*)p;        p += (size_t)M_TOTAL * 48 * sizeof(float);
  float* dl = (float*)p;        p += (size_t)M_TOTAL * D_INNER * sizeof(float);
  __bf16* xbf  = (__bf16*)p;    p += (size_t)M_TOTAL * D_MODEL * sizeof(__bf16);
  __bf16* ubf  = (__bf16*)p;    p += (size_t)M_TOTAL * D_INNER * sizeof(__bf16);
  __bf16* ybf  = (__bf16*)p;    p += (size_t)M_TOTAL * D_INNER * sizeof(__bf16);
  __bf16* inwb = (__bf16*)p;    p += (size_t)1024 * 256 * sizeof(__bf16);
  __bf16* xpwb = (__bf16*)p;    p += (size_t)48 * 512 * sizeof(__bf16);
  __bf16* owb  = (__bf16*)p;    p += (size_t)256 * 512 * sizeof(__bf16);

  // one-shot bf16 staging (exact grids; all sizes divisible by 256)
  k_cvt_bf16<<<(M_TOTAL * D_MODEL) / 256, 256, 0, stream>>>(x, xbf);
  k_cvt_bf16<<<(1024 * 256) / 256, 256, 0, stream>>>(inw, inwb);
  k_cvt_bf16<<<(48 * 512) / 256, 256, 0, stream>>>(xpw, xpwb);
  k_cvt_bf16<<<(256 * 512) / 256, 256, 0, stream>>>(ow, owb);

  k_in_proj<<<dim3(2, M_TOTAL / 16), 256, 0, stream>>>(xbf, inwb, xz);
  k_conv_silu<<<(M_TOTAL * D_INNER) / 256, 256, 0, stream>>>(xz, cw, cb, u, ubf);
  k_x_proj<<<M_TOTAL / 16, 96, 0, stream>>>(ubf, xpwb, xd);
  k_dt<<<(M_TOTAL * D_INNER) / 256, 256, 0, stream>>>(xd, dtw, dtb, dl);
  k_scan<<<BATCH * 2, 256, 0, stream>>>(xz, u, xd, dl, alog, dp, ybf);
  k_out_proj_ln<<<M_TOTAL / 16, 256, 0, stream>>>(ybf, owb, lng, lnb, out);
}
